// FNNAttenModel_16071767622130
// MI455X (gfx1250) — matrix operand order repaired and checked by reading, not compiled, not run on hardware
//
#include <hip/hip_runtime.h>

// ---------------------------------------------------------------------------
// Types for CDNA5 WMMA / TDM (gfx1250, wave32)
// ---------------------------------------------------------------------------
typedef __bf16 bf16;
typedef __attribute__((ext_vector_type(16))) __bf16 v16bf;
typedef __attribute__((ext_vector_type(8)))  float  v8f;
typedef __attribute__((ext_vector_type(4)))  unsigned int u32x4;
typedef __attribute__((ext_vector_type(8)))  int i32x8;
typedef __attribute__((ext_vector_type(4)))  int i32x4;

// Model dims
#define SEQ 128
#define BATCH 64
#define NINP 1024
#define NHID 1024
#define NTOKEN 32000
#define NHEADS 16
#define HEAD_DIM 64
#define MROWS (BATCH * SEQ)   // 8192

// ---------------------------------------------------------------------------
// Fragment loaders.
// A-matrix 16x32 bf16 layout: lane L, m = L%16, kb = (L/16)*16;
//   element e -> K = kb + e   (16 consecutive K values per lane)
// B-matrix 32x16 bf16 layout: lane L, n = L%16, kb = (L/16)*16;
//   element e -> K = kb + e   (16 consecutive K values per lane)
// ---------------------------------------------------------------------------
__device__ __forceinline__ v16bf load_a_frag(const bf16* p) {
  union { v16bf v; uint4 q[2]; } u;
  u.q[0] = *(const uint4*)(p);        // K = kb .. kb+7
  u.q[1] = *(const uint4*)(p + 8);    // K = kb+8 .. kb+15
  return u.v;
}
__device__ __forceinline__ v16bf load_b_frag_bf16(const bf16* p) {
  union { v16bf v; uint4 q[2]; } u;
  u.q[0] = *(const uint4*)(p);        // 16 consecutive bf16
  u.q[1] = *(const uint4*)(p + 8);
  return u.v;
}
__device__ __forceinline__ v16bf cvt_b_frag_f32(const float* p) {
  v16bf v;
#pragma unroll
  for (int e = 0; e < 16; ++e) v[e] = (bf16)p[e];  // fp32 -> bf16 in regs
  return v;
}

// ---------------------------------------------------------------------------
// Tensor Data Mover: 2D tile load Global -> LDS.
// D# per ISA 8.3/8.4. Groups 2/3 zero (tile_dim2/3 = 0 -> dims unused).
// data_size = 1 (2-byte elements). Dims/strides in element units.
// 6-arg builtin form: (g0 u32x4, g1 i32x8, g2 i32x4, g3 i32x4, i32x8, cpol)
// ---------------------------------------------------------------------------
__device__ __forceinline__ void tdm_load_2d_bf16(
    unsigned lds_byte_off, const void* gptr,
    unsigned tensor_d0, unsigned tensor_d1,
    unsigned tile_d0, unsigned tile_d1, unsigned long stride0_elems) {
  const unsigned long ga = (unsigned long)gptr;
  u32x4 g0;
  g0[0] = 1u;                                        // count=1, user mode
  g0[1] = lds_byte_off;                              // lds_addr (bytes)
  g0[2] = (unsigned)(ga & 0xFFFFFFFFu);              // global_addr[31:0]
  g0[3] = (unsigned)((ga >> 32) & 0x01FFFFFFu)       // global_addr[56:32]
          | (2u << 30);                              // type = 2 ("image")
  i32x8 g1;
  g1[0] = (int)(1u << 16);                           // wg_mask=0, data_size=1 (2B)
  g1[1] = (int)((tensor_d0 & 0xFFFFu) << 16);        // abar=0 | tensor_dim0[15:0]
  g1[2] = (int)(((tensor_d0 >> 16) & 0xFFFFu) | ((tensor_d1 & 0xFFFFu) << 16));
  g1[3] = (int)(((tensor_d1 >> 16) & 0xFFFFu) | ((tile_d0 & 0xFFFFu) << 16));
  g1[4] = (int)(tile_d1 & 0xFFFFu);                  // tile_dim1 | tile_dim2=0
  g1[5] = (int)(stride0_elems & 0xFFFFFFFFu);        // dim0_stride[31:0]
  g1[6] = (int)((stride0_elems >> 32) & 0xFFFFu);    // dim0_stride[47:32] | dim1_stride[15:0]=0
  g1[7] = 0;
  i32x4 z4 = {0, 0, 0, 0};
  i32x8 z8 = {0, 0, 0, 0, 0, 0, 0, 0};
  __builtin_amdgcn_tensor_load_to_lds(g0, g1, z4, z4, z8, 0);
}

// ---------------------------------------------------------------------------
// Embedding gather + fp32 -> bf16 convert.
// emb[(b*SEQ+s), d] = emb_W[input[s*BATCH+b], d]
// ---------------------------------------------------------------------------
__global__ void gather_embed(const int* __restrict__ input,
                             const float* __restrict__ embW,
                             bf16* __restrict__ out) {
  int m = blockIdx.x;            // 0 .. 8191, m = b*SEQ + s
  int b = m >> 7;
  int s = m & 127;
  int tok = input[s * BATCH + b];
  const float* src = embW + (long)tok * NINP;
  bf16* dst = out + (long)m * NINP;
  for (int d = threadIdx.x; d < NINP; d += blockDim.x) dst[d] = (bf16)src[d];
}

// ---------------------------------------------------------------------------
// Generic WMMA GEMM:  out[M,N] = A_bf16[M,K] @ W_f32[N,K]^T (+ bias)
// One wave computes a (16*MT) x 16 output tile. K is covered by
// gridDim.y slices of (kchunks * 32) for split-K accumulation.
// OUT_MODE: 0 = bf16 store (+bias), 1 = f32 store (+bias),
//           3 = f32 partial store (no bias), slice-major [gridDim.y][M][ldo]
// ---------------------------------------------------------------------------
template <int MT, int OUT_MODE>
__global__ __launch_bounds__(128) void gemm_wmma(
    const bf16* __restrict__ A, const float* __restrict__ W,
    const float* __restrict__ bias, void* __restrict__ out,
    int M, int N, int K, int kchunks, int ldo) {
  const int wavesPerBlock = blockDim.x >> 5;
  const int wave = blockIdx.x * wavesPerBlock + (threadIdx.x >> 5);
  const int lane = threadIdx.x & 31;
  const int half = lane >> 4;
  const int l16  = lane & 15;

  const int ntiles = N >> 4;
  const int n_tile = wave % ntiles;
  const int m_super = wave / ntiles;
  if (m_super * (16 * MT) >= M) return;

  const int k0 = blockIdx.y * (kchunks << 5);
  const long outBase = (long)blockIdx.y * (long)M * (long)ldo;  // 0 unless split-K

  v8f zero = {};
  v8f acc[MT];
#pragma unroll
  for (int mt = 0; mt < MT; ++mt) acc[mt] = zero;

  const int nCol = (n_tile << 4) + l16;
  const long wrow = (long)nCol * (long)K;

  for (int kc = 0; kc < kchunks; ++kc) {
    const int kk = k0 + (kc << 5);
    const float* wp = W + wrow + kk + (half << 4);
    // speculative prefetch of next weight K-chunk (global_prefetch_b8)
    __builtin_prefetch(wp + 32, 0, 1);
    v16bf bF = cvt_b_frag_f32(wp);
#pragma unroll
    for (int mt = 0; mt < MT; ++mt) {
      const int row = m_super * (16 * MT) + (mt << 4) + l16;
      v16bf aF = load_a_frag(A + (long)row * K + kk + (half << 4));
      acc[mt] = __builtin_amdgcn_wmma_f32_16x16x32_bf16(
          false, aF, false, bF, (short)0, acc[mt], false, false);
    }
  }

  const float bv = bias ? bias[nCol] : 0.0f;
#pragma unroll
  for (int mt = 0; mt < MT; ++mt) {
#pragma unroll
    for (int r = 0; r < 8; ++r) {
      const int m = m_super * (16 * MT) + (mt << 4) + r + (half << 3);
      const float v = acc[mt][r] + bv;
      if (OUT_MODE == 0) {
        ((bf16*)out)[(long)m * ldo + nCol] = (bf16)v;
      } else if (OUT_MODE == 1) {
        ((float*)out)[(long)m * ldo + nCol] = v;
      } else {  // 3: split-K partial
        ((float*)out)[outBase + (long)m * ldo + nCol] = v;
      }
    }
  }
}

// ---------------------------------------------------------------------------
// Attention: one workgroup (8 waves) per (batch, head).
// scores = Q K^T / 4 ; softmax ; out = P V
// Q,K 128x64 bf16 staged via Tensor Data Mover; V transposed manually
// (TDM cannot transpose). P 128x128 bf16 in LDS (wave-private strips).
// ---------------------------------------------------------------------------
__global__ __launch_bounds__(256) void attention_wmma(
    const bf16* __restrict__ Qb, const bf16* __restrict__ Kb,
    const bf16* __restrict__ Vb, bf16* __restrict__ Xb) {
  extern __shared__ __align__(16) char smem[];
  bf16* sQ  = (bf16*)smem;                  // 128*64   LDS offset 0
  bf16* sK  = sQ + SEQ * HEAD_DIM;          // 128*64   LDS offset 16384
  bf16* sVt = sK + SEQ * HEAD_DIM;          // 64*128   LDS offset 32768
  bf16* sP  = sVt + HEAD_DIM * SEQ;         // 128*128  LDS offset 49152

  const int bh = blockIdx.x;
  const int b  = bh >> 4;
  const int h  = bh & 15;
  const long base = ((long)b * SEQ) * NINP + h * HEAD_DIM;  // + s*NINP + d
  const int tid = threadIdx.x;

  // Wave 0 kicks off TDM: two 128x64 bf16 tiles (row stride NINP elements).
  if (tid < 32) {
    tdm_load_2d_bf16(0u,     Qb + base, HEAD_DIM, SEQ, HEAD_DIM, SEQ, NINP);
    tdm_load_2d_bf16(16384u, Kb + base, HEAD_DIM, SEQ, HEAD_DIM, SEQ, NINP);
  }
  // Meanwhile all 256 threads stage V transposed (d-major).
  for (int i = tid; i < SEQ * HEAD_DIM; i += 256) {
    const int s = i >> 6, d = i & 63;
    sVt[d * SEQ + s] = Vb[base + (long)s * NINP + d];
  }
  if (tid < 32) __builtin_amdgcn_s_wait_tensorcnt(0);
  __syncthreads();

  const int wave = tid >> 5, lane = tid & 31;
  const int half = lane >> 4, l16 = lane & 15;
  const int q0 = wave * 16;

  // Q fragments for the two 32-wide K chunks of head_dim=64
  v16bf aQ[2];
#pragma unroll
  for (int kc = 0; kc < 2; ++kc)
    aQ[kc] = load_a_frag(sQ + (q0 + l16) * HEAD_DIM + kc * 32 + half * 16);

  v8f zero = {};
  v8f accS[8];
#pragma unroll
  for (int t = 0; t < 8; ++t) accS[t] = zero;

#pragma unroll
  for (int t = 0; t < 8; ++t) {
#pragma unroll
    for (int kc = 0; kc < 2; ++kc) {
      v16bf bK = load_b_frag_bf16(sK + (t * 16 + l16) * HEAD_DIM + kc * 32 + half * 16);
      accS[t] = __builtin_amdgcn_wmma_f32_16x16x32_bf16(
          false, aQ[kc], false, bK, (short)0, accS[t], false, false);
    }
  }

  // softmax over the 128 columns; lane holds rows r + 8*half of this strip.
  const float scale = 0.25f;  // 1/sqrt(NHEADS)
  float rmax[8], rsum[8];
#pragma unroll
  for (int r = 0; r < 8; ++r) {
    float m = -3.4e38f;
#pragma unroll
    for (int t = 0; t < 8; ++t) m = fmaxf(m, accS[t][r]);
    for (int off = 1; off < 16; off <<= 1) m = fmaxf(m, __shfl_xor(m, off, 32));
    rmax[r] = m * scale;
  }
#pragma unroll
  for (int r = 0; r < 8; ++r) {
    float s = 0.0f;
#pragma unroll
    for (int t = 0; t < 8; ++t) {
      const float p = __expf(accS[t][r] * scale - rmax[r]);
      accS[t][r] = p;
      s += p;
    }
    for (int off = 1; off < 16; off <<= 1) s += __shfl_xor(s, off, 32);
    rsum[r] = s;
  }
  // P strip (wave-private) to LDS so it can be re-fragmented as an A operand.
#pragma unroll
  for (int t = 0; t < 8; ++t)
#pragma unroll
    for (int r = 0; r < 8; ++r)
      sP[(q0 + r + half * 8) * SEQ + t * 16 + l16] = (bf16)(accS[t][r] / rsum[r]);

  // out strip 16x64 = P[16x128] @ V[128x64]
  v8f accO[4];
#pragma unroll
  for (int dt = 0; dt < 4; ++dt) accO[dt] = zero;
#pragma unroll
  for (int kc = 0; kc < 4; ++kc) {
    v16bf aP = load_a_frag(sP + (q0 + l16) * SEQ + kc * 32 + half * 16);
#pragma unroll
    for (int dt = 0; dt < 4; ++dt) {
      v16bf bV = load_b_frag_bf16(sVt + (dt * 16 + l16) * SEQ + kc * 32 + half * 16);
      accO[dt] = __builtin_amdgcn_wmma_f32_16x16x32_bf16(
          false, aP, false, bV, (short)0, accO[dt], false, false);
    }
  }
#pragma unroll
  for (int dt = 0; dt < 4; ++dt)
#pragma unroll
    for (int r = 0; r < 8; ++r) {
      const int q = q0 + r + half * 8;
      const int d = dt * 16 + l16;
      Xb[base + (long)q * NINP + d] = (bf16)accO[dt][r];
    }
}

// ---------------------------------------------------------------------------
// Split-K reduction + bias + tanh for the hidden layer.
// ---------------------------------------------------------------------------
__global__ void reduce_tanh(const float* __restrict__ partial,
                            const float* __restrict__ bias,
                            bf16* __restrict__ h, int slices, int MN) {
  const int i = blockIdx.x * blockDim.x + threadIdx.x;
  if (i >= MN) return;
  float s = 0.0f;
  for (int k = 0; k < slices; ++k) s += partial[(long)k * MN + i];
  h[i] = (bf16)tanhf(s + bias[i & (NHID - 1)]);
}

// ---------------------------------------------------------------------------
// Launcher
// ---------------------------------------------------------------------------
extern "C" void kernel_launch(void* const* d_in, const int* in_sizes, int n_in,
                              void* d_out, int out_size, void* d_ws, size_t ws_size,
                              hipStream_t stream) {
  const int*   input = (const int*)d_in[0];
  const float* embW  = (const float*)d_in[1];
  const float* wq = (const float*)d_in[2];  const float* bq = (const float*)d_in[3];
  const float* wk = (const float*)d_in[4];  const float* bk = (const float*)d_in[5];
  const float* wv = (const float*)d_in[6];  const float* bv = (const float*)d_in[7];
  const float* wo = (const float*)d_in[8];  const float* bo = (const float*)d_in[9];
  const float* wh = (const float*)d_in[10]; const float* bh = (const float*)d_in[11];
  const float* wd = (const float*)d_in[12]; const float* bd = (const float*)d_in[13];
  float* out = (float*)d_out;

  // Workspace layout (bf16 activations, heavy region reuse): 4 x 16MB + 128KB
  char* ws = (char*)d_ws;
  const size_t SZ16 = (size_t)MROWS * NINP * sizeof(bf16);  // 16 MB
  bf16* emb = (bf16*)(ws);             // embeddings, later attention output
  bf16* Qb  = (bf16*)(ws + SZ16);      // Q, later wo output ("flat")
  bf16* Kb  = (bf16*)(ws + 2 * SZ16);  // K, later wh split-K partials (f32)
  bf16* Vb  = (bf16*)(ws + 3 * SZ16);  // V
  bf16* hB  = (bf16*)(ws + 4 * SZ16);  // hidden activations [64,1024]

  // 1) embedding gather + convert
  gather_embed<<<MROWS, 256, 0, stream>>>(input, embW, emb);

  // 2) Q/K/V projections: [8192,1024] @ [1024,1024]^T
  //    waves = (1024/16) * (8192/16) = 32768 -> 8192 blocks of 4 waves
  gemm_wmma<1, 0><<<dim3(8192), 128, 0, stream>>>(emb, wq, bq, Qb, MROWS, NINP, NINP, 32, NINP);
  gemm_wmma<1, 0><<<dim3(8192), 128, 0, stream>>>(emb, wk, bk, Kb, MROWS, NINP, NINP, 32, NINP);
  gemm_wmma<1, 0><<<dim3(8192), 128, 0, stream>>>(emb, wv, bv, Vb, MROWS, NINP, NINP, 32, NINP);

  // 3) attention: 1024 (b,h) workgroups, 80 KB dynamic LDS each
  attention_wmma<<<BATCH * NHEADS, 256, 81920, stream>>>(Qb, Kb, Vb, emb);

  // 4) output projection -> flat [64, 131072] bf16 (same memory layout)
  gemm_wmma<1, 0><<<dim3(8192), 128, 0, stream>>>(emb, wo, bo, Qb, MROWS, NINP, NINP, 32, NINP);

  // 5) hidden layer: [64,131072] @ [131072,1024]^T, split-K over 64 slices of 2048
  float* partial = (float*)(ws + 2 * SZ16);  // 64*64*1024*4 = 16 MB
  gemm_wmma<4, 3><<<dim3(16, 64), 128, 0, stream>>>((const bf16*)Qb, wh, nullptr, partial,
                                                    BATCH, NHID, SEQ * NINP, 64, NHID);
  reduce_tanh<<<(BATCH * NHID + 255) / 256, 256, 0, stream>>>(partial, bh, hB, 64, BATCH * NHID);

  // 6) decoder: [64,1024] @ [1024,32000]^T -> fp32 out
  //    waves = (32000/16) * 1 = 2000 -> 500 blocks
  gemm_wmma<4, 1><<<dim3(500), 128, 0, stream>>>(hB, wd, bd, out, BATCH, NTOKEN, NINP, 32, NTOKEN);
}
